// TimeAwareMultiHeadAttention_42812234006866
// MI455X (gfx1250) — matrix operand order repaired and checked by reading, not compiled, not run on hardware
//
#include <hip/hip_runtime.h>

#define B_  8
#define L_  256
#define D_  128
#define H_  4
#define HD_ 32

typedef __attribute__((ext_vector_type(16))) _Float16 v16h;
typedef __attribute__((ext_vector_type(8)))  float    v8f;
typedef __attribute__((ext_vector_type(4)))  float    v4f;

// CDNA5 wave32 WMMA f16 A-operand K index for half j (0..15), lane-group g (0..1).
// lanes0-15: K=0..15 ; lanes16-31: K=16..31 (contiguous, matches B)
__device__ __forceinline__ int a_kidx(int j, int g) {
    return j + 16 * g;
}

// ---------------------------------------------------------------------------
// Kernel A: fused projections.
//   sel=0: q    = queries@Wq + bq              -> qbuf  [B,H,L,HD] f16
//   sel=1: kpk  = keys@Wk + bk + abs_pos_K     -> kpkT  [B,H,HD,L] f16 (transposed)
//   sel=2: vpv  = keys@Wv + bv + abs_pos_V     -> vpvb  [B,H,L,HD] f16
// grid (128 row-tiles, 3), block 256 (8 waves, one 16-col tile per wave)
// ---------------------------------------------------------------------------
__global__ __launch_bounds__(256)
void proj_kernel(const float* __restrict__ queries,
                 const float* __restrict__ keys,
                 const float* __restrict__ Wq, const float* __restrict__ bq,
                 const float* __restrict__ Wk, const float* __restrict__ bk,
                 const float* __restrict__ Wv, const float* __restrict__ bv,
                 const float* __restrict__ apK, const float* __restrict__ apV,
                 _Float16* __restrict__ qbuf,
                 _Float16* __restrict__ kpkT,
                 _Float16* __restrict__ vpvb)
{
    __shared__ _Float16 xs[16][D_];         // 16 x 128 activation strip, f16

    const int rt  = blockIdx.x;             // rows rt*16 .. rt*16+15 of B*L
    const int sel = blockIdx.y;
    const int t   = threadIdx.x;

    const float* X    = (sel == 0) ? queries : keys;
    const float* W    = (sel == 0) ? Wq : (sel == 1 ? Wk : Wv);
    const float* bias = (sel == 0) ? bq : (sel == 1 ? bk : bv);

    // stage activation strip to LDS (f32 -> f16), 2048 elems, 8 per thread
    {
        const float* src = X + (size_t)rt * 16 * D_;
        #pragma unroll
        for (int u = 0; u < 8; ++u) {
            int idx = t + u * 256;
            ((_Float16*)xs)[idx] = (_Float16)src[idx];
        }
    }
    __syncthreads();

    const int wave = t >> 5;
    const int lane = t & 31;
    const int g    = lane >> 4;
    const int n    = lane & 15;
    const int n0   = wave * 16;             // output column tile

    v8f c = {};
    for (int kk = 0; kk < 4; ++kk) {        // K = 128 in chunks of 32
        const int kbase = kk * 32;
        v16h a, bm;
        #pragma unroll
        for (int j = 0; j < 16; ++j) {
            a[j]  = xs[n][kbase + a_kidx(j, g)];
            bm[j] = (_Float16)W[(size_t)(kbase + j + 16 * g) * D_ + (n0 + n)];
        }
        c = __builtin_amdgcn_wmma_f32_16x16x32_f16(false, a, false, bm,
                                                   (short)0, c, false, false);
    }

    #pragma unroll
    for (int r = 0; r < 8; ++r) {
        const int mi  = r + 8 * g;          // row in tile (C layout: M = r + 8g)
        const int row = rt * 16 + mi;
        const int bb  = row / L_;
        const int ll  = row % L_;
        const int col = n0 + n;
        const int hh  = col / HD_;
        const int dd  = col % HD_;
        float val = c[r] + bias[col];
        if (sel == 1) {
            val += apK[(size_t)row * D_ + col];
            kpkT[(((size_t)bb * H_ + hh) * HD_ + dd) * L_ + ll] = (_Float16)val;
        } else if (sel == 2) {
            val += apV[(size_t)row * D_ + col];
            vpvb[(((size_t)bb * H_ + hh) * L_ + ll) * HD_ + dd] = (_Float16)val;
        } else {
            qbuf[(((size_t)bb * H_ + hh) * L_ + ll) * HD_ + dd] = (_Float16)val;
        }
    }
}

// ---------------------------------------------------------------------------
// Kernel B: scores = (q@kpk^T + tK.q) * 1/sqrt(HD), mask, softmax -> w (f16)
// grid (L/16, H, B), block 256 (8 waves; wave w owns m-tiles 2w, 2w+1)
// tK is a 256 MB read-once stream -> non-temporal loads (keep L2 for reused data)
// ---------------------------------------------------------------------------
__global__ __launch_bounds__(256)
void score_softmax_kernel(const _Float16* __restrict__ qbuf,
                          const _Float16* __restrict__ kpkT,
                          const float* __restrict__ tK,
                          const int* __restrict__ pad,     // padding_mask [B,L]
                          _Float16* __restrict__ wbuf)
{
    __shared__ float S[16][L_];      // 16 KB score tile
    __shared__ float qs[16][HD_];    // q tile in f32 for the tK dot pass
    __shared__ float red[16][16];
    __shared__ float red2[16][16];

    const int lt = blockIdx.x, h = blockIdx.y, b = blockIdx.z;
    const int l0 = lt * 16;
    const int t  = threadIdx.x;
    const int wave = t >> 5, lane = t & 31, g = lane >> 4, n = lane & 15;

    const size_t qb = (((size_t)b * H_ + h) * L_ + l0) * HD_;

    // stage q tile as f32 (512 elems)
    #pragma unroll
    for (int u = 0; u < 2; ++u) {
        int idx = t + u * 256;
        ((float*)qs)[idx] = (float)qbuf[qb + idx];
    }

    // WMMA q @ kpk^T : A = 16x32 (K = HD, exactly one WMMA per m-tile)
    v16h a;
    #pragma unroll
    for (int j = 0; j < 16; ++j)
        a[j] = qbuf[qb + (size_t)n * HD_ + a_kidx(j, g)];

    const size_t kb = (((size_t)b * H_ + h) * HD_) * L_;
    for (int mt = 2 * wave; mt < 2 * wave + 2; ++mt) {
        v16h bm;
        #pragma unroll
        for (int j = 0; j < 16; ++j)
            bm[j] = kpkT[kb + (size_t)(j + 16 * g) * L_ + mt * 16 + n];
        v8f c = {};
        c = __builtin_amdgcn_wmma_f32_16x16x32_f16(false, a, false, bm,
                                                   (short)0, c, false, false);
        #pragma unroll
        for (int r = 0; r < 8; ++r)
            S[r + 8 * g][mt * 16 + n] = c[r];
    }
    __syncthreads();

    // tK pass: thread == m; coalesced non-temporal float4 stream of tK
    const int   m     = t;
    const float scale = 0.17677669529663687f;       // 1/sqrt(32)
    const float NEGV  = -4294967295.0f;             // -2^32 + 1
    for (int l = 0; l < 16; ++l) {
        const v4f* p = (const v4f*)(tK +
            ((((size_t)b * L_ + l0 + l) * L_ + m) * D_ + h * HD_));
        float dot = 0.f;
        #pragma unroll
        for (int u = 0; u < 8; ++u) {
            v4f v4 = __builtin_nontemporal_load(p + u);
            dot = fmaf(v4.x, qs[l][4 * u + 0], dot);
            dot = fmaf(v4.y, qs[l][4 * u + 1], dot);
            dot = fmaf(v4.z, qs[l][4 * u + 2], dot);
            dot = fmaf(v4.w, qs[l][4 * u + 3], dot);
        }
        float s = (S[l][m] + dot) * scale;
        const bool masked = (pad[b * L_ + l0 + l] != 0) || (m > l0 + l);
        S[l][m] = masked ? NEGV : s;
    }
    __syncthreads();

    // softmax: 16 threads per row
    const int row = t >> 4, j = t & 15;
    float mx = -3.402823466e38f;
    for (int mm = j; mm < L_; mm += 16) mx = fmaxf(mx, S[row][mm]);
    red[row][j] = mx;
    __syncthreads();
    float rmax = red[row][0];
    #pragma unroll
    for (int u = 1; u < 16; ++u) rmax = fmaxf(rmax, red[row][u]);
    float sum = 0.f;
    for (int mm = j; mm < L_; mm += 16) {
        float e = __expf(S[row][mm] - rmax);
        S[row][mm] = e;
        sum += e;
    }
    red2[row][j] = sum;
    __syncthreads();
    float rsum = 0.f;
    #pragma unroll
    for (int u = 0; u < 16; ++u) rsum += red2[row][u];
    const float inv = 1.0f / rsum;
    _Float16* wrow = wbuf + (((size_t)b * H_ + h) * L_ + l0 + row) * L_;
    for (int mm = j; mm < L_; mm += 16)
        wrow[mm] = (_Float16)(S[row][mm] * inv);
}

// ---------------------------------------------------------------------------
// Kernel C: out = w @ (v+pv)  [WMMA]  +  sum_m w[l,m]*tV[b,l,m,h,:]  [NT stream]
// grid (L/16, H, B), block 256
// ---------------------------------------------------------------------------
__global__ __launch_bounds__(256)
void out_kernel(const _Float16* __restrict__ wbuf,
                const _Float16* __restrict__ vpvb,
                const float* __restrict__ tV,
                float* __restrict__ out)
{
    __shared__ _Float16 ws[16][L_];   // 8 KB weight tile
    __shared__ float    acc[16][HD_]; // 2 KB WMMA result

    const int lt = blockIdx.x, h = blockIdx.y, b = blockIdx.z;
    const int l0 = lt * 16;
    const int t  = threadIdx.x;
    const int wave = t >> 5, lane = t & 31, g = lane >> 4, n = lane & 15;

    // stage 16x256 weight tile (contiguous)
    {
        const _Float16* src = wbuf + (((size_t)b * H_ + h) * L_ + l0) * L_;
        #pragma unroll
        for (int u = 0; u < 16; ++u)
            ((_Float16*)ws)[t + u * 256] = src[t + u * 256];
    }
    __syncthreads();

    // w @ vpv : M=16, N=32 (two 16-tiles on waves 0,1), K=256 in 8 chunks
    if (wave < 2) {
        const int n0 = wave * 16;
        const size_t vb = (((size_t)b * H_ + h) * L_) * HD_;
        v8f c = {};
        for (int kk = 0; kk < 8; ++kk) {
            v16h a, bm;
            #pragma unroll
            for (int j = 0; j < 16; ++j) {
                a[j]  = ws[n][kk * 32 + a_kidx(j, g)];
                bm[j] = vpvb[vb + (size_t)(kk * 32 + j + 16 * g) * HD_ + n0 + n];
            }
            c = __builtin_amdgcn_wmma_f32_16x16x32_f16(false, a, false, bm,
                                                       (short)0, c, false, false);
        }
        #pragma unroll
        for (int r = 0; r < 8; ++r)
            acc[r + 8 * g][n0 + n] = c[r];
    }

    // tV weighted reduction: lane d in [0,32), group handles l = grp and grp+8;
    // 256 MB read-once stream -> non-temporal loads
    const int d   = t & 31;
    const int grp = t >> 5;
    float tacc[2] = {0.f, 0.f};
    #pragma unroll
    for (int li = 0; li < 2; ++li) {
        const int l = grp + li * 8;
        const float* tvp = tV +
            ((((size_t)b * L_ + l0 + l) * L_) * D_ + h * HD_ + d);
        float s_ = 0.f;
        for (int m = 0; m < L_; ++m)
            s_ = fmaf((float)ws[l][m],
                      __builtin_nontemporal_load(tvp + (size_t)m * D_), s_);
        tacc[li] = s_;
    }
    __syncthreads();

    #pragma unroll
    for (int li = 0; li < 2; ++li) {
        const int l = grp + li * 8;
        out[((size_t)b * L_ + l0 + l) * D_ + h * HD_ + d] = acc[l][d] + tacc[li];
    }
}

// ---------------------------------------------------------------------------
extern "C" void kernel_launch(void* const* d_in, const int* in_sizes, int n_in,
                              void* d_out, int out_size, void* d_ws, size_t ws_size,
                              hipStream_t stream)
{
    const float* queries = (const float*)d_in[0];
    const float* keys    = (const float*)d_in[1];
    const int*   pad     = (const int*)d_in[2];   // padding_mask [B,L,1]
    // d_in[3] attn_mask: causal (~tril) — recomputed in-kernel as (m > l)
    const float* apK = (const float*)d_in[4];
    const float* apV = (const float*)d_in[5];
    const float* tK  = (const float*)d_in[6];
    const float* tV  = (const float*)d_in[7];
    const float* Wq  = (const float*)d_in[8];
    const float* bq  = (const float*)d_in[9];
    const float* Wk  = (const float*)d_in[10];
    const float* bk  = (const float*)d_in[11];
    const float* Wv  = (const float*)d_in[12];
    const float* bv  = (const float*)d_in[13];

    char* ws = (char*)d_ws;
    _Float16* qbuf = (_Float16*)ws; ws += (size_t)B_ * H_ * L_ * HD_ * 2; // 512 KB
    _Float16* kpkT = (_Float16*)ws; ws += (size_t)B_ * H_ * L_ * HD_ * 2; // 512 KB
    _Float16* vpvb = (_Float16*)ws; ws += (size_t)B_ * H_ * L_ * HD_ * 2; // 512 KB
    _Float16* wbuf = (_Float16*)ws;                                       // 4 MB

    proj_kernel<<<dim3((B_ * L_) / 16, 3), 256, 0, stream>>>(
        queries, keys, Wq, bq, Wk, bk, Wv, bv, apK, apV, qbuf, kpkT, vpvb);
    score_softmax_kernel<<<dim3(L_ / 16, H_, B_), 256, 0, stream>>>(
        qbuf, kpkT, tK, pad, wbuf);
    out_kernel<<<dim3(L_ / 16, H_, B_), 256, 0, stream>>>(
        wbuf, vpvb, tV, (float*)d_out);
}
